// QuantumAttentionLayer_927712936474
// MI455X (gfx1250) — compile-verified, matrix operand order checked by reading, not run on hardware
//
#include <hip/hip_runtime.h>
#include <hip/hip_bf16.h>
#include <math.h>

// ---------------- problem constants (from reference) ----------------
#define HDIM   768
#define SEQ    2048
#define BATCH  8
#define NTOK   (BATCH * SEQ)
#define DIM16  16          // 2^4 quantum state dim
#define NLAYER 2
#define NCHUNK (SEQ / 32)
#define VSTRIDE 772        // 768 + 4 pad words: 772 % 64 != 0 -> no LDS bank aliasing

typedef _Float16 v16h __attribute__((ext_vector_type(16)));
typedef float    v8f  __attribute__((ext_vector_type(8)));

union H16 { v16h v; _Float16 e[16]; uint32_t u[8]; };
union F8  { v8f  v; float    e[8];  };

__device__ __forceinline__ float2 cmul(float2 a, float2 b) {
    return make_float2(a.x * b.x - a.y * b.y, a.x * b.y + a.y * b.x);
}
__device__ __forceinline__ float2 cadd(float2 a, float2 b) {
    return make_float2(a.x + b.x, a.y + b.y);
}

// pack two f32 into adjacent f16 halves of a u32 (lowers to v_cvt_pk_f16_f32)
__device__ __forceinline__ uint32_t pkh2(float lo, float hi) {
    union { _Float16 h[2]; uint32_t u; } t;
    t.h[0] = (_Float16)lo;
    t.h[1] = (_Float16)hi;
    return t.u;
}

// K-index map shared by the 16-bit A(16x32) and B(32x16) WMMA fragments
// (ISA 7.12.2): vgpr v<4 -> K=2v, v>=4 -> K=16+2(v-4); lanes>=16 add +8.
__device__ __forceinline__ int wmma_kbase(int v, int half) {
    return ((v < 4) ? (2 * v) : (16 + 2 * (v - 4))) + 8 * half;
}
// pair index (K/2) for the packed-LDS layout
__device__ __forceinline__ int wmma_kpair(int v, int half) {
    return ((v < 4) ? v : (8 + (v - 4))) + 4 * half;
}

// ---------------- kernel 1: build 16x16 VQC unitary ----------------
__global__ void build_unitary_kernel(const float* __restrict__ w,
                                     float2* __restrict__ Uout) {
    if (threadIdx.x != 0 || blockIdx.x != 0) return;
    float2 U[DIM16][DIM16];
    for (int i = 0; i < DIM16; ++i)
        for (int j = 0; j < DIM16; ++j)
            U[i][j] = make_float2(i == j ? 1.f : 0.f, 0.f);

    for (int l = 0; l < NLAYER; ++l) {
        // single-qubit Rot matrices: RZ(om) RY(th) RZ(phi)
        float2 R[4][2][2];
        for (int q = 0; q < 4; ++q) {
            float phi = w[l * 12 + q * 3 + 0];
            float th  = w[l * 12 + q * 3 + 1];
            float om  = w[l * 12 + q * 3 + 2];
            float ch = cosf(0.5f * th), sh = sinf(0.5f * th);
            float P = 0.5f * (phi + om), M = 0.5f * (phi - om);
            R[q][0][0] = make_float2( ch * cosf(P), -ch * sinf(P)); // exp(-iP)c
            R[q][0][1] = make_float2(-sh * cosf(M), -sh * sinf(M)); // -exp(iM)s
            R[q][1][0] = make_float2( sh * cosf(M), -sh * sinf(M)); // exp(-iM)s
            R[q][1][1] = make_float2( ch * cosf(P),  ch * sinf(P)); // exp(iP)c
        }
        // Mm = kron(R0..R3) @ U   (wire 0 = MSB)
        float2 Mm[DIM16][DIM16];
        for (int i = 0; i < DIM16; ++i)
            for (int j = 0; j < DIM16; ++j)
                Mm[i][j] = make_float2(0.f, 0.f);
        for (int i = 0; i < DIM16; ++i) {
            for (int k = 0; k < DIM16; ++k) {
                float2 rf = make_float2(1.f, 0.f);
                for (int q = 0; q < 4; ++q)
                    rf = cmul(rf, R[q][(i >> (3 - q)) & 1][(k >> (3 - q)) & 1]);
                for (int j = 0; j < DIM16; ++j)
                    Mm[i][j] = cadd(Mm[i][j], cmul(rf, U[k][j]));
            }
        }
        // entangler: sequential CNOT(i, (i+r)%4) acting on basis states
        int r = (l % 3) + 1;
        for (int b = 0; b < DIM16; ++b) {
            int b2 = b;
            for (int i2 = 0; i2 < 4; ++i2)
                if ((b2 >> (3 - i2)) & 1) b2 ^= 1 << (3 - ((i2 + r) & 3));
            for (int j = 0; j < DIM16; ++j) U[b2][j] = Mm[b][j];
        }
    }
    for (int i = 0; i < DIM16; ++i)
        for (int j = 0; j < DIM16; ++j)
            Uout[i * DIM16 + j] = U[i][j];
}

// ---------------- per-token VQC evaluation (whole wave, uniform x) ----------------
__device__ __forceinline__ void vqc_eval(const float2* __restrict__ U,
                                         const float x[4], float z[4]) {
    float c[4], s[4];
#pragma unroll
    for (int q = 0; q < 4; ++q) { c[q] = cosf(0.5f * x[q]); s[q] = sinf(0.5f * x[q]); }
    int i = threadIdx.x & 15; // lanes 16..31 duplicate lanes 0..15
    float2 psi = make_float2(0.f, 0.f);
#pragma unroll
    for (int j = 0; j < DIM16; ++j) {
        float2 p = (j & 8) ? make_float2(0.f, -s[0]) : make_float2(c[0], 0.f);
        p = cmul(p, (j & 4) ? make_float2(0.f, -s[1]) : make_float2(c[1], 0.f));
        p = cmul(p, (j & 2) ? make_float2(0.f, -s[2]) : make_float2(c[2], 0.f));
        p = cmul(p, (j & 1) ? make_float2(0.f, -s[3]) : make_float2(c[3], 0.f));
        psi = cadd(psi, cmul(U[i * DIM16 + j], p));
    }
    float prob = psi.x * psi.x + psi.y * psi.y;
    float t0 = (i & 8) ? -prob : prob;
    float t1 = (i & 4) ? -prob : prob;
    float t2 = (i & 2) ? -prob : prob;
    float t3 = (i & 1) ? -prob : prob;
#pragma unroll
    for (int m = 1; m < 16; m <<= 1) {
        t0 += __shfl_xor(t0, m);
        t1 += __shfl_xor(t1, m);
        t2 += __shfl_xor(t2, m);
        t3 += __shfl_xor(t3, m);
    }
    z[0] = t0; z[1] = t1; z[2] = t2; z[3] = t3;
}

// ---------------- kernel 2: Q/K projection + VQC -> Qv, Kv ----------------
__global__ void __launch_bounds__(256)
proj_vqc_kernel(const float* __restrict__ emb,
                const float* __restrict__ Wq, const float* __restrict__ bq,
                const float* __restrict__ Wk, const float* __restrict__ bk,
                const float2* __restrict__ U,
                float* __restrict__ Qv, float* __restrict__ Kv) {
    int wave = threadIdx.x >> 5;
    int lane = threadIdx.x & 31;
    int token = blockIdx.x * 8 + wave;
    const float* e = emb + (size_t)token * HDIM;

    float aq[4] = {0.f, 0.f, 0.f, 0.f};
    float ak[4] = {0.f, 0.f, 0.f, 0.f};
    for (int h = lane; h < HDIM; h += 32) {
        float ev = e[h];
#pragma unroll
        for (int q = 0; q < 4; ++q) {
            aq[q] = fmaf(ev, Wq[q * HDIM + h], aq[q]);
            ak[q] = fmaf(ev, Wk[q * HDIM + h], ak[q]);
        }
    }
#pragma unroll
    for (int q = 0; q < 4; ++q) {
#pragma unroll
        for (int off = 16; off; off >>= 1) {
            aq[q] += __shfl_xor(aq[q], off);
            ak[q] += __shfl_xor(ak[q], off);
        }
    }
    float xq[4], xk[4];
#pragma unroll
    for (int q = 0; q < 4; ++q) { xq[q] = aq[q] + bq[q]; xk[q] = ak[q] + bk[q]; }

    float zq[4], zk[4];
    vqc_eval(U, xq, zq);
    vqc_eval(U, xk, zk);
    if (lane == 0) {
#pragma unroll
        for (int q = 0; q < 4; ++q) {
            Qv[(size_t)token * 4 + q] = zq[q];
            Kv[(size_t)token * 4 + q] = zk[q];
        }
    }
}

// ---------------- kernel 3: flash attention, P@V on WMMA f16 ----------------
// grid: (SEQ/16, BATCH), block 256 (8 waves). Wave w owns 16 rows x cols [w*96, w*96+96).
// Double-buffered: LDS holds pair-packed f16 V chunks (Vt[kp][col] = {V[2kp][c], V[2kp+1][c]})
// so each WMMA B fragment is 8 ds_load_b32; V is staged with coalesced float4 loads.
__global__ void __launch_bounds__(256)
attn_kernel(const float* __restrict__ emb,
            const float* __restrict__ Qv, const float* __restrict__ Kv,
            float* __restrict__ out) {
    __shared__ uint32_t Vt[2][16 * VSTRIDE];   // 2 x 48.25 KB packed f16 pairs
    __shared__ _Float16 pTile[2][16][32];      // exp-score tiles
    __shared__ float dred[16][16];
    __shared__ float rdenom[16];

    const int tid  = threadIdx.x;
    const int lane = tid & 31;
    const int wave = tid >> 5;
    const int half = lane >> 4;
    const int mn   = lane & 15;
    const int b    = blockIdx.y;
    const int row0 = blockIdx.x * 16;

    // staging / score roles: r = query-row or k-pair index, jo = col group
    const int r  = tid >> 4;
    const int jo = tid & 15;

    const float4* Qv4 = reinterpret_cast<const float4*>(Qv);
    const float4* Kv4 = reinterpret_cast<const float4*>(Kv);
    const float4 qrow = Qv4[(size_t)b * SEQ + row0 + r];
    const float* embB = emb + (size_t)b * SEQ * HDIM;

    float dsum = 0.f;
    F8 acc[6];
#pragma unroll
    for (int t = 0; t < 6; ++t) acc[t].v = (v8f){0.f, 0.f, 0.f, 0.f, 0.f, 0.f, 0.f, 0.f};

    // ---- prologue: stage chunk 0 into buffer 0 ----
    {
        float4 kv1 = Kv4[(size_t)b * SEQ + jo];
        float4 kv2 = Kv4[(size_t)b * SEQ + 16 + jo];
        float e1 = __expf(qrow.x * kv1.x + qrow.y * kv1.y + qrow.z * kv1.z + qrow.w * kv1.w);
        float e2 = __expf(qrow.x * kv2.x + qrow.y * kv2.y + qrow.z * kv2.z + qrow.w * kv2.w);
        dsum += e1 + e2;
        pTile[0][r][jo]      = (_Float16)e1;
        pTile[0][r][16 + jo] = (_Float16)e2;

        const float* rA = embB + (size_t)(2 * r) * HDIM;
        const float* rB = rA + HDIM;
#pragma unroll
        for (int i = 0; i < 12; ++i) {
            const int c = jo * 4 + 64 * i;
            float4 a4 = *reinterpret_cast<const float4*>(rA + c);
            float4 b4 = *reinterpret_cast<const float4*>(rB + c);
            uint4 pk;
            pk.x = pkh2(a4.x, b4.x); pk.y = pkh2(a4.y, b4.y);
            pk.z = pkh2(a4.z, b4.z); pk.w = pkh2(a4.w, b4.w);
            *reinterpret_cast<uint4*>(&Vt[0][r * VSTRIDE + c]) = pk;
        }
    }
    __syncthreads();

    for (int chunk = 0; chunk < NCHUNK; ++chunk) {
        const int cur = chunk & 1;
        const int nxt = cur ^ 1;
        const bool do_stage = (chunk + 1 < NCHUNK);

        // ---- issue next chunk's global loads + scores first (latency hides under WMMA) ----
        float4 a4[12], b4[12];
        const float* rA = embB;
        if (do_stage) {
            const int jb2 = (chunk + 1) * 32;
            float4 kv1 = Kv4[(size_t)b * SEQ + jb2 + jo];
            float4 kv2 = Kv4[(size_t)b * SEQ + jb2 + 16 + jo];
            float e1 = __expf(qrow.x * kv1.x + qrow.y * kv1.y + qrow.z * kv1.z + qrow.w * kv1.w);
            float e2 = __expf(qrow.x * kv2.x + qrow.y * kv2.y + qrow.z * kv2.z + qrow.w * kv2.w);
            dsum += e1 + e2;
            pTile[nxt][r][jo]      = (_Float16)e1;
            pTile[nxt][r][16 + jo] = (_Float16)e2;

            rA = embB + (size_t)(jb2 + 2 * r) * HDIM;
            const float* rB = rA + HDIM;
#pragma unroll
            for (int i = 0; i < 12; ++i) {
                const int c = jo * 4 + 64 * i;
                a4[i] = *reinterpret_cast<const float4*>(rA + c);
                b4[i] = *reinterpret_cast<const float4*>(rB + c);
            }
            if (chunk + 2 < NCHUNK)
                __builtin_prefetch(rA + 32 * HDIM + jo * 4, 0, 1);
        }

        // ---- compute current chunk: A frag from pTile, B frags from packed Vt ----
        H16 af;
#pragma unroll
        for (int v = 0; v < 8; ++v)
            af.u[v] = *reinterpret_cast<const uint32_t*>(&pTile[cur][mn][wmma_kbase(v, half)]);
#pragma unroll
        for (int t = 0; t < 6; ++t) {
            const int col = wave * 96 + t * 16 + mn;
            H16 bf;
#pragma unroll
            for (int v = 0; v < 8; ++v)
                bf.u[v] = Vt[cur][wmma_kpair(v, half) * VSTRIDE + col];
            acc[t].v = __builtin_amdgcn_wmma_f32_16x16x32_f16(
                false, af.v, false, bf.v, (short)0, acc[t].v, false, false);
        }

        // ---- pack + store next chunk's V into the other buffer ----
        if (do_stage) {
#pragma unroll
            for (int i = 0; i < 12; ++i) {
                const int c = jo * 4 + 64 * i;
                uint4 pk;
                pk.x = pkh2(a4[i].x, b4[i].x); pk.y = pkh2(a4[i].y, b4[i].y);
                pk.z = pkh2(a4[i].z, b4[i].z); pk.w = pkh2(a4[i].w, b4[i].w);
                *reinterpret_cast<uint4*>(&Vt[nxt][r * VSTRIDE + c]) = pk;
            }
        }
        __syncthreads();
    }

    // ---- softmax denominator: reduce per-row partial sums ----
    dred[r][jo] = dsum;
    __syncthreads();
    if (tid < 16) {
        float ssum = 0.f;
#pragma unroll
        for (int cix = 0; cix < 16; ++cix) ssum += dred[tid][cix];
        rdenom[tid] = 1.0f / ssum;
    }
    __syncthreads();

    // ---- scale + store (C layout: vgpr rr -> M = rr + 8*half, N = mn) ----
#pragma unroll
    for (int t = 0; t < 6; ++t) {
        const int col = wave * 96 + t * 16 + mn;
#pragma unroll
        for (int rr = 0; rr < 8; ++rr) {
            const int m = rr + 8 * half;
            out[((size_t)b * SEQ + row0 + m) * HDIM + col] = acc[t].e[rr] * rdenom[m];
        }
    }
}

// ---------------- launcher ----------------
extern "C" void kernel_launch(void* const* d_in, const int* in_sizes, int n_in,
                              void* d_out, int out_size, void* d_ws, size_t ws_size,
                              hipStream_t stream) {
    const float* emb = (const float*)d_in[0];
    const float* Wq  = (const float*)d_in[1];
    const float* bq  = (const float*)d_in[2];
    const float* Wk  = (const float*)d_in[3];
    const float* bk  = (const float*)d_in[4];
    const float* vw  = (const float*)d_in[5];
    float* out = (float*)d_out;

    // workspace layout
    char* ws = (char*)d_ws;
    float2* U  = (float2*)ws;                       // 256 * 8B   = 2 KB
    float*  Qv = (float*)(ws + 2048);               // 16384*4*4B = 256 KB
    float*  Kv = (float*)(ws + 2048 + (size_t)NTOK * 4 * sizeof(float));

    build_unitary_kernel<<<1, 32, 0, stream>>>(vw, U);
    proj_vqc_kernel<<<NTOK / 8, 256, 0, stream>>>(emb, Wq, bq, Wk, bk, U, Qv, Kv);
    dim3 grid(SEQ / 16, BATCH);
    attn_kernel<<<grid, 256, 0, stream>>>(emb, Qv, Kv, out);
}
